// MultiHeadAttention_20916490731986
// MI455X (gfx1250) — compile-verified, matrix operand order checked by reading, not run on hardware
//
#include <hip/hip_runtime.h>
#include <math.h>

// ---------------- constants ----------------
#define BDIM   8
#define SDIM   1024
#define HIDD   1024
#define ADIM   1024
#define HEADS  16
#define SUBD   64
#define DEXT   128              // extended head dim: qh(64) || lh(64)
#define BHN    (BDIM*HEADS)     // 128 batch-heads

typedef __bf16 bf16_t;
typedef __attribute__((ext_vector_type(16))) __bf16 v16bf;
typedef __attribute__((ext_vector_type(8)))  float  v8f;
typedef unsigned int uintv4 __attribute__((ext_vector_type(4)));
typedef int          intv4  __attribute__((ext_vector_type(4)));
typedef int          intv8  __attribute__((ext_vector_type(8)));

// ---------------- helpers ----------------
static __device__ __forceinline__ bf16_t f2bf(float f) {
  unsigned u = __builtin_bit_cast(unsigned, f);
  unsigned r = u + 0x7FFFu + ((u >> 16) & 1u);   // round-to-nearest-even
  unsigned short h = (unsigned short)(r >> 16);
  return __builtin_bit_cast(bf16_t, h);
}

union FragU { v16bf v; uint4 u[2]; };

// Load a 16x32 bf16 A/B fragment (ISA 16-bit A-matrix layout, wave32):
// lane L: row/col = L&15 ; element e: K = k0 + 16*(e>>3) + 8*(L>>4) + (e&7)
// => two contiguous 16B chunks per lane. Works for generic (global or LDS) ptrs.
static __device__ __forceinline__ v16bf load_frag(const bf16_t* base, int ld, int row, int k0) {
  const int lane = threadIdx.x & 31;
  const int hf   = lane >> 4;
  const int ln   = lane & 15;
  const bf16_t* p = base + (size_t)(row + ln) * ld + k0 + 8 * hf;
  FragU f;
  f.u[0] = *(const uint4*)(p);
  f.u[1] = *(const uint4*)(p + 16);
  return f.v;
}

static __device__ __forceinline__ v8f wmma_bf16(v16bf a, v16bf b, v8f c) {
  return __builtin_amdgcn_wmma_f32_16x16x32_bf16(false, a, false, b, (short)0, c, false, false);
}

static __device__ __forceinline__ v8f zero8() {
  v8f z = {0.f,0.f,0.f,0.f,0.f,0.f,0.f,0.f};
  return z;
}

// 2-stage ping-pong pipelined 16x64 GEMM tile: C[4] += A[rbase:+16, :] * BT[cbase:+64, :]^T
// No register copies: buffers A/B alternate roles each half-iteration.
template <int K>
static __device__ __forceinline__ void gemm_tile(
    const bf16_t* __restrict__ A, const bf16_t* __restrict__ BT,
    int rbase, int cbase, v8f C[4])
{
  v16bf afA = load_frag(A, K, rbase, 0);
  v16bf bfA[4];
  #pragma unroll
  for (int nf = 0; nf < 4; nf++) bfA[nf] = load_frag(BT, K, cbase + nf * 16, 0);

  for (int k0 = 0; k0 < K; k0 += 64) {     // K is a multiple of 64
    v16bf afB = load_frag(A, K, rbase, k0 + 32);
    v16bf bfB[4];
    #pragma unroll
    for (int nf = 0; nf < 4; nf++) bfB[nf] = load_frag(BT, K, cbase + nf * 16, k0 + 32);
    #pragma unroll
    for (int nf = 0; nf < 4; nf++) C[nf] = wmma_bf16(afA, bfA[nf], C[nf]);

    const int kn = (k0 + 64 < K) ? (k0 + 64) : 0;   // last iter: harmless reload
    afA = load_frag(A, K, rbase, kn);
    #pragma unroll
    for (int nf = 0; nf < 4; nf++) bfA[nf] = load_frag(BT, K, cbase + nf * 16, kn);
    #pragma unroll
    for (int nf = 0; nf < 4; nf++) C[nf] = wmma_bf16(afB, bfB[nf], C[nf]);
  }
}

// Issue a TDM load of a 2D tile (rows x rowUnits*8B) into LDS (TENSORcnt-tracked).
static __device__ __forceinline__ void tdm_load_tile(
    const void* gptr, unsigned ldsOff, int rowUnits, int rows, int strideUnits)
{
  unsigned long long ga = (unsigned long long)(uintptr_t)gptr;
  uintv4 g0;
  g0.x = 1u;                                            // count=1, user mode
  g0.y = ldsOff;                                        // lds_addr
  g0.z = (unsigned)(ga & 0xffffffffu);                  // global_addr[31:0]
  g0.w = (unsigned)((ga >> 32) & 0x01ffffffu) | (2u << 30);  // addr[56:32] | type=2
  intv8 g1;
  g1[0] = (int)(3u << 16);                              // data_size=3 (8B units)
  g1[1] = (int)((unsigned)rowUnits << 16);              // tensor_dim0[15:0]
  g1[2] = (int)((unsigned)rows << 16);                  // tensor_dim1[15:0] (dim0 hi = 0)
  g1[3] = (int)((unsigned)rowUnits << 16);              // tile_dim0 (dim1 hi = 0)
  g1[4] = rows;                                         // tile_dim1 ; tile_dim2=0
  g1[5] = strideUnits;                                  // tensor_dim0_stride[31:0]
  g1[6] = 0;
  g1[7] = 0;
  intv4 gz = {0, 0, 0, 0};
#if defined(__clang_major__) && (__clang_major__ >= 23)
  intv8 g4 = {0, 0, 0, 0, 0, 0, 0, 0};
  __builtin_amdgcn_tensor_load_to_lds(g0, g1, gz, gz, g4, 0);
#else
  __builtin_amdgcn_tensor_load_to_lds(g0, g1, gz, gz, 0);
#endif
}

// ---------------- prep kernels ----------------
__global__ void zero_f32_kernel(float* __restrict__ p, int n) {
  int i = blockIdx.x * 256 + threadIdx.x;
  if (i < n) p[i] = 0.f;
}

__global__ void cvt_f32_bf16_kernel(const float* __restrict__ src, bf16_t* __restrict__ dst, int n) {
  int i = blockIdx.x * 256 + threadIdx.x;
  if (i < n) dst[i] = f2bf(src[i]);
}

// dst (N x K, row-major bf16) = transpose of src (K x N, row-major f32)
__global__ void transpose_f32_bf16_kernel(const float* __restrict__ src, bf16_t* __restrict__ dst,
                                          int K, int N) {
  int i = blockIdx.x * 256 + threadIdx.x;
  if (i < K * N) {
    int n = i / K, k = i % K;
    dst[i] = f2bf(src[(size_t)k * N + n]);
  }
}

// ---------------- projection GEMM: C = relu(A @ B + bias), head-split epilogue ---------
// A: (8192 x K) bf16 row-major ; BT: (1024 x K) bf16 row-major (= B transposed)
// mode 0: -> Qe[bh, s, 0..63]   mode 1: -> Ke[bh, s, 0..63]
// mode 2: -> Qe & Ke [bh, s, 64..127]   mode 3: -> Vt[bh, d, s]
template <int K>
__global__ void __launch_bounds__(128) proj_gemm_kernel(
    const bf16_t* __restrict__ A, const bf16_t* __restrict__ BT,
    const float* __restrict__ bias, int mode,
    bf16_t* __restrict__ Qe, bf16_t* __restrict__ Ke, bf16_t* __restrict__ Vt)
{
  const int wave = threadIdx.x >> 5;
  const int lane = threadIdx.x & 31;
  const int hf   = lane >> 4;
  const int ln   = lane & 15;
  const int rbase = blockIdx.y * 64 + wave * 16;
  const int cbase = blockIdx.x * 64;

  v8f C[4];
  #pragma unroll
  for (int nf = 0; nf < 4; nf++) C[nf] = zero8();

  gemm_tile<K>(A, BT, rbase, cbase, C);

  #pragma unroll
  for (int nf = 0; nf < 4; nf++) {
    const int col = cbase + nf * 16 + ln;
    const float bv = bias[col];
    const int hh = col >> 6;          // head
    const int d  = col & 63;          // dim in head
    #pragma unroll
    for (int r = 0; r < 8; r++) {
      const int row = rbase + r + 8 * hf;   // C layout: M = r + 8*(lane>>4)
      const int b = row >> 10, s = row & 1023;
      const int bh = b * HEADS + hh;
      float v = fmaxf(C[nf][r] + bv, 0.f);
      bf16_t w = f2bf(v);
      if (mode == 0)      Qe[((size_t)bh * SDIM + s) * DEXT + d] = w;
      else if (mode == 1) Ke[((size_t)bh * SDIM + s) * DEXT + d] = w;
      else if (mode == 2) {
        Qe[((size_t)bh * SDIM + s) * DEXT + 64 + d] = w;
        Ke[((size_t)bh * SDIM + s) * DEXT + 64 + d] = w;
      } else              Vt[((size_t)bh * SUBD + d) * SDIM + s] = w;
    }
  }
}

// ---------------- flash attention per batch-head ----------------
// Qe,Ke: (BH, S, 128) bf16 ; Vt: (BH, 64, S) bf16 ; yB: (B, S, A) bf16
// Ke key-block (128x128 bf16 = 32KB) is staged into LDS once per WG via the
// Tensor Data Mover (TENSORcnt), then all 4 waves read fragments from LDS.
__global__ void __launch_bounds__(128) attn_kernel(
    const bf16_t* __restrict__ Qe, const bf16_t* __restrict__ Ke,
    const bf16_t* __restrict__ Vt, const int* __restrict__ mask,
    bf16_t* __restrict__ yB)
{
  __shared__ __align__(16) unsigned short ldsKe[128 * 128];    // 32 KB key tile
  __shared__ __align__(16) unsigned short ldsP[4 * 16 * 136];  // per-wave 16x128 P tile (pad 136)
  const int bh    = blockIdx.y;
  const int qbase = blockIdx.x * 64;
  const int wave  = threadIdx.x >> 5;
  const int lane  = threadIdx.x & 31;
  const int hf    = lane >> 4;
  const int ln    = lane & 15;
  const int rbase = qbase + wave * 16;

  const bf16_t* Qp  = Qe + (size_t)bh * SDIM * DEXT;
  const bf16_t* Kp  = Ke + (size_t)bh * SDIM * DEXT;
  const bf16_t* Vp  = Vt + (size_t)bh * SUBD * SDIM;
  const int*   mrow = mask + (size_t)(bh % BDIM) * SDIM;  // reference's head-major mask quirk

  v16bf qf[4];
  #pragma unroll
  for (int kk = 0; kk < 4; kk++) qf[kk] = load_frag(Qp, DEXT, rbase, kk * 32);

  v8f O[4];
  #pragma unroll
  for (int nf = 0; nf < 4; nf++) O[nf] = zero8();
  float m[8], l[8];
  #pragma unroll
  for (int r = 0; r < 8; r++) { m[r] = -__builtin_inff(); l[r] = 0.f; }

  unsigned short* myP = &ldsP[wave * 16 * 136];
  const bf16_t* Kl = (const bf16_t*)ldsKe;
  const unsigned ldsKeOff = (unsigned)(uintptr_t)(void*)ldsKe;
  const float LOG2E = 1.4426950408889634f;

  for (int kb = 0; kb < SDIM; kb += 128) {
    __syncthreads();                       // all waves done reading previous Ke tile
    if (wave == 0) {
      // TDM: 128 rows x 256B (= 32 x 8B units per row), contiguous rows
      tdm_load_tile(Kp + (size_t)kb * DEXT, ldsKeOff, /*rowUnits=*/32, /*rows=*/128,
                    /*strideUnits=*/32);
      __builtin_amdgcn_s_wait_tensorcnt(0);
    }
    __syncthreads();                       // Ke tile visible to all waves

    // ---- S = (Q·K^T) * 0.125, masked ----
    v8f sf[8];
    #pragma unroll
    for (int f = 0; f < 8; f++) {
      v8f s = zero8();
      #pragma unroll
      for (int kk = 0; kk < 4; kk++) {
        v16bf bfr = load_frag(Kl, DEXT, f * 16, kk * 32);
        s = wmma_bf16(qf[kk], bfr, s);
      }
      const int mv = mrow[kb + f * 16 + ln];   // per-key mask (col depends only on lane)
      #pragma unroll
      for (int r = 0; r < 8; r++) {
        float x = s[r] * 0.125f;
        sf[f][r] = (mv == 1) ? -1.0e30f : x;
      }
    }
    // ---- row max over block (8 frags in-reg, then 16-lane shfl tree) ----
    float t[8];
    #pragma unroll
    for (int r = 0; r < 8; r++) {
      float v = sf[0][r];
      #pragma unroll
      for (int f = 1; f < 8; f++) v = fmaxf(v, sf[f][r]);
      t[r] = v;
    }
    #pragma unroll
    for (int w = 1; w < 16; w <<= 1) {
      #pragma unroll
      for (int r = 0; r < 8; r++) t[r] = fmaxf(t[r], __shfl_xor(t[r], w, 32));
    }
    float alpha[8];
    #pragma unroll
    for (int r = 0; r < 8; r++) {
      float mn = fmaxf(m[r], t[r]);
      alpha[r] = exp2f((m[r] - mn) * LOG2E);
      m[r] = mn;
    }
    // ---- P = exp(S - m), row sums; stage P into LDS as bf16 (C->A layout) ----
    float rs[8];
    #pragma unroll
    for (int r = 0; r < 8; r++) rs[r] = 0.f;
    #pragma unroll
    for (int f = 0; f < 8; f++) {
      #pragma unroll
      for (int r = 0; r < 8; r++) {
        float p = exp2f((sf[f][r] - m[r]) * LOG2E);
        rs[r] += p;
        myP[(r + 8 * hf) * 136 + f * 16 + ln] = __builtin_bit_cast(unsigned short, f2bf(p));
      }
    }
    #pragma unroll
    for (int w = 1; w < 16; w <<= 1) {
      #pragma unroll
      for (int r = 0; r < 8; r++) rs[r] += __shfl_xor(rs[r], w, 32);
    }
    #pragma unroll
    for (int r = 0; r < 8; r++) l[r] = l[r] * alpha[r] + rs[r];
    #pragma unroll
    for (int nf = 0; nf < 4; nf++)
      #pragma unroll
      for (int r = 0; r < 8; r++) O[nf][r] *= alpha[r];

    asm volatile("s_wait_dscnt 0x0" ::: "memory");   // P stores -> P fragment loads (same wave)

    // ---- O += P @ V  (fully unrolled; V fragments prefetched one step ahead) ----
    const bf16_t* Pp = (const bf16_t*)myP;
    v16bf vcur[4];
    #pragma unroll
    for (int nf = 0; nf < 4; nf++) vcur[nf] = load_frag(Vp, SDIM, nf * 16, kb);
    #pragma unroll
    for (int kk = 0; kk < 4; kk++) {
      v16bf pf = load_frag(Pp, 136, 0, kk * 32);
      const int knext = kb + ((kk < 3) ? (kk + 1) : kk) * 32;
      v16bf vnext[4];
      #pragma unroll
      for (int nf = 0; nf < 4; nf++) vnext[nf] = load_frag(Vp, SDIM, nf * 16, knext);
      #pragma unroll
      for (int nf = 0; nf < 4; nf++) O[nf] = wmma_bf16(pf, vcur[nf], O[nf]);
      #pragma unroll
      for (int nf = 0; nf < 4; nf++) vcur[nf] = vnext[nf];
    }
  }

  // ---- normalize and write y merged-heads (B,S,A) as bf16 ----
  const int b  = bh >> 4;
  const int hh = bh & 15;
  #pragma unroll
  for (int r = 0; r < 8; r++) {
    const float inv = 1.0f / l[r];
    const int s = rbase + r + 8 * hf;
    #pragma unroll
    for (int nf = 0; nf < 4; nf++) {
      const int col = hh * SUBD + nf * 16 + ln;
      yB[((size_t)(b * SDIM + s)) * ADIM + col] = f2bf(O[nf][r] * inv);
    }
  }
}

// ---------------- output GEMM: o[b,:] = sum_s relu(y @ Wo + bo) ----------------
__global__ void __launch_bounds__(128) out_gemm_kernel(
    const bf16_t* __restrict__ A, const bf16_t* __restrict__ BT,
    const float* __restrict__ bias, float* __restrict__ out)
{
  const int wave = threadIdx.x >> 5;
  const int lane = threadIdx.x & 31;
  const int hf   = lane >> 4;
  const int ln   = lane & 15;
  const int rbase = blockIdx.y * 64 + wave * 16;
  const int cbase = blockIdx.x * 64;

  v8f C[4];
  #pragma unroll
  for (int nf = 0; nf < 4; nf++) C[nf] = zero8();

  gemm_tile<HIDD>(A, BT, rbase, cbase, C);

  const int b = rbase >> 10;   // 1024 rows (= S) per batch, 64-row tiles never cross b
  #pragma unroll
  for (int nf = 0; nf < 4; nf++) {
    const int col = cbase + nf * 16 + ln;
    const float bv = bias[col];
    float sum = 0.f;
    #pragma unroll
    for (int r = 0; r < 8; r++) sum += fmaxf(C[nf][r] + bv, 0.f);
    sum += __shfl_xor(sum, 16, 32);          // combine rows 0..7 and 8..15
    if (hf == 0) {
      atomicAdd(&out[(size_t)b * HIDD + col], sum);
      atomicAdd(&out[(size_t)(BDIM * HIDD) + (size_t)b * HIDD + col], sum); // duplicated output
    }
  }
}

// ---------------- host launcher ----------------
extern "C" void kernel_launch(void* const* d_in, const int* in_sizes, int n_in,
                              void* d_out, int out_size, void* d_ws, size_t ws_size,
                              hipStream_t stream) {
  (void)in_sizes; (void)n_in; (void)out_size; (void)ws_size;
  const float* h    = (const float*)d_in[0];
  const int*   mask = (const int*)d_in[1];
  // d_in[2] (g), d_in[10] (Wg), d_in[11] (bg): unused — the global term is a
  // per-(b,h) constant across the softmax axis and cancels exactly.
  const float* l  = (const float*)d_in[3];
  const float* Wq = (const float*)d_in[4];
  const float* bq = (const float*)d_in[5];
  const float* Wk = (const float*)d_in[6];
  const float* bk = (const float*)d_in[7];
  const float* Wv = (const float*)d_in[8];
  const float* bv = (const float*)d_in[9];
  const float* Wl = (const float*)d_in[12];
  const float* bl = (const float*)d_in[13];
  const float* Wo = (const float*)d_in[14];
  const float* bo = (const float*)d_in[15];
  float* out = (float*)d_out;

  char* ws = (char*)d_ws;
  size_t off = 0;
  auto take = [&](size_t bytes) -> char* {
    char* p = ws + off;
    off += (bytes + 255) & ~(size_t)255;
    return p;
  };
  bf16_t* hB  = (bf16_t*)take((size_t)BDIM * SDIM * HIDD * 2);  // 16 MB
  bf16_t* lB  = (bf16_t*)take((size_t)BDIM * SDIM * 128  * 2);  //  2 MB
  bf16_t* WqT = (bf16_t*)take((size_t)ADIM * HIDD * 2);
  bf16_t* WkT = (bf16_t*)take((size_t)ADIM * HIDD * 2);
  bf16_t* WvT = (bf16_t*)take((size_t)ADIM * HIDD * 2);
  bf16_t* WoT = (bf16_t*)take((size_t)HIDD * ADIM * 2);
  bf16_t* WlT = (bf16_t*)take((size_t)ADIM * 128  * 2);
  bf16_t* Qe  = (bf16_t*)take((size_t)BHN * SDIM * DEXT * 2);   // 32 MB
  bf16_t* Ke  = (bf16_t*)take((size_t)BHN * SDIM * DEXT * 2);   // 32 MB
  bf16_t* Vt  = (bf16_t*)take((size_t)BHN * SUBD * SDIM * 2);   // 16 MB
  bf16_t* yB  = hB;  // reuse: h not needed after projections (stream-ordered)

  // zero outputs (atomics accumulate into them)
  zero_f32_kernel<<<(2 * BDIM * HIDD + 255) / 256, 256, 0, stream>>>(out, 2 * BDIM * HIDD);

  // bf16 conversions / weight transposes
  cvt_f32_bf16_kernel<<<(BDIM * SDIM * HIDD) / 256, 256, 0, stream>>>(h, hB, BDIM * SDIM * HIDD);
  cvt_f32_bf16_kernel<<<(BDIM * SDIM * 128) / 256, 256, 0, stream>>>(l, lB, BDIM * SDIM * 128);
  transpose_f32_bf16_kernel<<<(HIDD * ADIM) / 256, 256, 0, stream>>>(Wq, WqT, HIDD, ADIM);
  transpose_f32_bf16_kernel<<<(HIDD * ADIM) / 256, 256, 0, stream>>>(Wk, WkT, HIDD, ADIM);
  transpose_f32_bf16_kernel<<<(HIDD * ADIM) / 256, 256, 0, stream>>>(Wv, WvT, HIDD, ADIM);
  transpose_f32_bf16_kernel<<<(ADIM * HIDD) / 256, 256, 0, stream>>>(Wo, WoT, ADIM, HIDD);
  transpose_f32_bf16_kernel<<<(128 * ADIM) / 256, 256, 0, stream>>>(Wl, WlT, 128, ADIM);

  // projections (fused bias+relu+head-split)
  dim3 pg(ADIM / 64, (BDIM * SDIM) / 64);   // (16, 128)
  proj_gemm_kernel<HIDD><<<pg, 128, 0, stream>>>(hB, WqT, bq, 0, Qe, Ke, Vt);
  proj_gemm_kernel<HIDD><<<pg, 128, 0, stream>>>(hB, WkT, bk, 1, Qe, Ke, Vt);
  proj_gemm_kernel<HIDD><<<pg, 128, 0, stream>>>(hB, WvT, bv, 3, Qe, Ke, Vt);
  proj_gemm_kernel<128> <<<pg, 128, 0, stream>>>(lB, WlT, bl, 2, Qe, Ke, Vt);

  // flash attention per batch-head (TDM-staged key tiles)
  attn_kernel<<<dim3(SDIM / 64, BHN), 128, 0, stream>>>(Qe, Ke, Vt, mask, yB);

  // output GEMM + relu + row-sum (atomics into both duplicated outputs)
  out_gemm_kernel<<<dim3(HIDD / 64, (BDIM * SDIM) / 64), 128, 0, stream>>>(yB, WoT, bo, out);
}
